// OCRLosses_10496900072267
// MI455X (gfx1250) — matrix operand order repaired and checked by reading, not compiled, not run on hardware
//
#include <hip/hip_runtime.h>
#include <hip/hip_bf16.h>

#define NUM_LABELS 1000
#define HIDDEN     512
#define BATCH      65536
#define NPAD       1008      // 63 * 16 column padding for W
#define NTILES     63
#define NOISE_STD  0.1f
#define MROWS      128       // M-panel rows per block (8 M-tiles)
#define XSTRIDE    520       // 512 + 8 halves pad -> conflict-free ds_load_b128

typedef _Float16 v16h __attribute__((ext_vector_type(16)));
typedef _Float16 v8h  __attribute__((ext_vector_type(8)));
typedef float    v8f  __attribute__((ext_vector_type(8)));
typedef float    v2f  __attribute__((ext_vector_type(2)));

// ---------------- workspace layout (bytes) ----------------
// [0,32)              : double AC[4]  {stab_sum, center_sum, brier_sum, sep_sum}
// [64, 64+4096)       : float counts[1024]
// [64+4096, 64+8192)  : float sq[1024]            (center row norms)
// then float sums[1000*512], float cent[1000*512], _Float16 Wt[1008*512]
#define ZERO_FLOATS (16 + 1024 + 1024 + NUM_LABELS * HIDDEN)

__device__ __forceinline__ float waveReduceSum(float v) {
#pragma unroll
  for (int m = 16; m >= 1; m >>= 1) v += __shfl_xor(v, m, 32);
  return v;
}

// K0: zero accumulators / counts / sq / sums
__global__ void k_zero(float* ws) {
  int t = blockIdx.x * blockDim.x + threadIdx.x;
  if (t < ZERO_FLOATS) ws[t] = 0.0f;
}

// K1: segment sums + counts (float4 per thread, atomics into L2)
__global__ void k_segsum(const float* __restrict__ cls, const int* __restrict__ labels,
                         float* __restrict__ sums, float* __restrict__ counts) {
  int t = blockIdx.x * blockDim.x + threadIdx.x;          // BATCH * 128 threads
  if (t >= BATCH * (HIDDEN / 4)) return;
  int b  = t >> 7;
  int h  = (t & 127) << 2;
  int lab = labels[b];
  float4 v = *reinterpret_cast<const float4*>(cls + (size_t)b * HIDDEN + h);
  float* s = sums + (size_t)lab * HIDDEN + h;
  atomicAdd(s + 0, v.x); atomicAdd(s + 1, v.y);
  atomicAdd(s + 2, v.z); atomicAdd(s + 3, v.w);
  if (h == 0) atomicAdd(counts + lab, 1.0f);
}

// K2: finalize centers (mean or passthrough) + accumulate row norms sq[c]
__global__ void k_centers(const float* __restrict__ sums, const float* __restrict__ counts,
                          const float* __restrict__ centers_in,
                          float* __restrict__ cent, float* __restrict__ sq) {
  int t = blockIdx.x * blockDim.x + threadIdx.x;          // NUM_LABELS * HIDDEN
  if (t >= NUM_LABELS * HIDDEN) return;
  int c = t / HIDDEN;
  float cnt = counts[c];
  float v = (cnt > 0.0f) ? (sums[t] / cnt) : centers_in[t];
  cent[t] = v;
  atomicAdd(sq + c, v * v);
}

// K2b: W (512x1000 f32, k-major) -> Wt (1008x512 f16, n-major, zero padded)
__global__ void k_wt(const float* __restrict__ W, _Float16* __restrict__ Wt) {
  int t = blockIdx.x * blockDim.x + threadIdx.x;          // NPAD * HIDDEN
  if (t >= NPAD * HIDDEN) return;
  int n = t / HIDDEN;
  int k = t % HIDDEN;
  float v = (n < NUM_LABELS) ? W[(size_t)k * NUM_LABELS + n] : 0.0f;
  Wt[t] = (_Float16)v;
}

// K3: fused  x = cls + 0.1*noise -> f16 LDS panel;  center-loss on the fly;
//     WMMA GEMM vs Wt; stability MSE vs logits. 256 thr = 8 waves, 128-row panel.
__global__ __launch_bounds__(256) void k_gemm_stab(
    const float* __restrict__ cls, const float* __restrict__ noise,
    const int* __restrict__ labels, const float* __restrict__ logits,
    const float* __restrict__ bias, const _Float16* __restrict__ Wt,
    const float* __restrict__ cent, double* __restrict__ AC) {
  __shared__ _Float16 xh[MROWS][XSTRIDE];
  __shared__ int labs[MROWS];
  const int tid  = threadIdx.x;
  const int row0 = blockIdx.x * MROWS;

  if (tid < MROWS) labs[tid] = labels[row0 + tid];
  __syncthreads();

  // ---- stage A panel (f16) + fused center loss ----
  float csum = 0.0f;
  for (int idx = tid; idx < MROWS * (HIDDEN / 4); idx += 256) {
    int r = idx >> 7;                 // HIDDEN/4 == 128
    int c = (idx & 127) << 2;
    size_t g = (size_t)(row0 + r) * HIDDEN + c;
    float4 cv = *reinterpret_cast<const float4*>(cls + g);
    float4 nv = *reinterpret_cast<const float4*>(noise + g);
    float4 ce = *reinterpret_cast<const float4*>(cent + (size_t)labs[r] * HIDDEN + c);
    float d0 = cv.x - ce.x, d1 = cv.y - ce.y, d2 = cv.z - ce.z, d3 = cv.w - ce.w;
    csum += d0 * d0 + d1 * d1 + d2 * d2 + d3 * d3;
    xh[r][c + 0] = (_Float16)(cv.x + NOISE_STD * nv.x);
    xh[r][c + 1] = (_Float16)(cv.y + NOISE_STD * nv.y);
    xh[r][c + 2] = (_Float16)(cv.z + NOISE_STD * nv.z);
    xh[r][c + 3] = (_Float16)(cv.w + NOISE_STD * nv.w);
  }
  __syncthreads();

  const int wave  = tid >> 5;
  const int lane  = tid & 31;
  const int lhalf = lane >> 4;     // 0: lanes 0-15, 1: lanes 16-31
  const int l15   = lane & 15;

  float ssum = 0.0f;
  for (int nt = wave; nt < NTILES; nt += 8) {
    const int ncol   = nt * 16 + l15;
    const int ncol_c = ncol < NUM_LABELS ? ncol : NUM_LABELS - 1;
    const bool valid = ncol < NUM_LABELS;
    const _Float16* wp = Wt + (size_t)ncol * HIDDEN;
    if (nt + 8 < NTILES)  // stream next strip of W through L2
      __builtin_prefetch(Wt + (size_t)((nt + 8) * 16 + l15) * HIDDEN, 0, 1);

    v8f acc[8];
#pragma unroll
    for (int m = 0; m < 8; ++m)
#pragma unroll
      for (int e = 0; e < 8; ++e) acc[m][e] = 0.0f;

    for (int ks = 0; ks < HIDDEN / 32; ++ks) {
      // B fragment: lane holds col=ncol, K = ks*32 + lhalf*8 + {0..7, 16..23}
      const int kb = ks * 32 + lhalf * 8;
      v8h blo = *reinterpret_cast<const v8h*>(wp + kb);
      v8h bhi = *reinterpret_cast<const v8h*>(wp + kb + 16);
      v16h bf;
#pragma unroll
      for (int e = 0; e < 8; ++e) { bf[e] = blo[e]; bf[e + 8] = bhi[e]; }

      // A fragments: lane holds row m*16+l15, K = ks*32 + lhalf*8 + {0..7, 16..23}
      const int ka = ks * 32 + lhalf * 8;
#pragma unroll
      for (int m = 0; m < 8; ++m) {
        const _Float16* xr = &xh[m * 16 + l15][0];
        v8h alo = *reinterpret_cast<const v8h*>(xr + ka);
        v8h ahi = *reinterpret_cast<const v8h*>(xr + ka + 16);
        v16h af;
#pragma unroll
        for (int e = 0; e < 8; ++e) { af[e] = alo[e]; af[e + 8] = ahi[e]; }
        acc[m] = __builtin_amdgcn_wmma_f32_16x16x32_f16(
            false, af, false, bf, (short)0, acc[m], false, false);
      }
    }

    // ---- epilogue: + bias, squared error vs clean logits ----
    const float bb = bias[ncol_c];
#pragma unroll
    for (int m = 0; m < 8; ++m) {
#pragma unroll
      for (int vg = 0; vg < 8; ++vg) {
        int row = row0 + m * 16 + lhalf * 8 + vg;
        float v = acc[m][vg] + bb;
        float lg = logits[(size_t)row * NUM_LABELS + ncol_c];
        float d = v - lg;
        ssum += valid ? d * d : 0.0f;
      }
    }
  }

  ssum = waveReduceSum(ssum);
  csum = waveReduceSum(csum);
  if (lane == 0) {
    atomicAdd(&AC[0], (double)ssum);
    atomicAdd(&AC[1], (double)csum);
  }
}

// K4: brier per row: softmax via LDS-staged row, sum p^2 - 2 p[label] + 1
__global__ __launch_bounds__(256) void k_brier(const float* __restrict__ logits,
                                               const int* __restrict__ labels,
                                               double* __restrict__ AC) {
  __shared__ float srow[NUM_LABELS];
  __shared__ float red[256];
  const int row = blockIdx.x, tid = threadIdx.x;
  float m = -3.4e38f;
  for (int i = tid; i < NUM_LABELS; i += 256) {
    float v = logits[(size_t)row * NUM_LABELS + i];
    srow[i] = v;
    m = fmaxf(m, v);
  }
  red[tid] = m; __syncthreads();
  for (int s = 128; s > 0; s >>= 1) { if (tid < s) red[tid] = fmaxf(red[tid], red[tid + s]); __syncthreads(); }
  const float M = red[0]; __syncthreads();

  float s1 = 0.0f, s2 = 0.0f;
  for (int i = tid; i < NUM_LABELS; i += 256) {
    float e = __expf(srow[i] - M);
    s1 += e; s2 += e * e;
  }
  red[tid] = s1; __syncthreads();
  for (int s = 128; s > 0; s >>= 1) { if (tid < s) red[tid] += red[tid + s]; __syncthreads(); }
  const float Z = red[0]; __syncthreads();
  red[tid] = s2; __syncthreads();
  for (int s = 128; s > 0; s >>= 1) { if (tid < s) red[tid] += red[tid + s]; __syncthreads(); }
  const float S2 = red[0];

  if (tid == 0) {
    float plab = __expf(srow[labels[row]] - M) / Z;
    float br = S2 / (Z * Z) - 2.0f * plab + 1.0f;
    atomicAdd(&AC[2], (double)br);
  }
}

// K5: separation Gram via f32 WMMA (16x16x4), then exp(-dist + I) sum
__global__ __launch_bounds__(32) void k_sep(const float* __restrict__ cent,
                                            const float* __restrict__ sq,
                                            double* __restrict__ AC) {
  const int ti = blockIdx.x, tj = blockIdx.y;
  const int lane = threadIdx.x;
  const int lhalf = lane >> 4, l15 = lane & 15;
  const int ra = ti * 16 + l15, cb = tj * 16 + l15;
  const int ra_c = ra < NUM_LABELS ? ra : NUM_LABELS - 1;
  const int cb_c = cb < NUM_LABELS ? cb : NUM_LABELS - 1;
  const float amask = (ra < NUM_LABELS) ? 1.0f : 0.0f;
  const float bmask = (cb < NUM_LABELS) ? 1.0f : 0.0f;
  const float* pa = cent + (size_t)ra_c * HIDDEN;
  const float* pb = cent + (size_t)cb_c * HIDDEN;

  v8f acc;
#pragma unroll
  for (int e = 0; e < 8; ++e) acc[e] = 0.0f;

  for (int k = 0; k < HIDDEN / 4; ++k) {
    // A 16x4: lanes 0-15 K={4k,4k+1}, lanes 16-31 K={4k+2,4k+3}
    const int ks = k * 4 + lhalf * 2;
    v2f a, b;
    a[0] = pa[ks] * amask;     a[1] = pa[ks + 1] * amask;
    b[0] = pb[ks] * bmask;     b[1] = pb[ks + 1] * bmask;
    acc = __builtin_amdgcn_wmma_f32_16x16x4_f32(
        false, a, false, b, (short)0, acc, false, false);
  }

  float s = 0.0f;
#pragma unroll
  for (int vg = 0; vg < 8; ++vg) {
    int i = ti * 16 + lhalf * 8 + vg;
    int j = tj * 16 + l15;
    if (i < NUM_LABELS && j < NUM_LABELS) {
      float d2 = sq[i] + sq[j] - 2.0f * acc[vg];
      d2 = fmaxf(d2, 0.0f);
      s += __expf(-__fsqrt_rn(d2) + ((i == j) ? 1.0f : 0.0f));
    }
  }
  s = waveReduceSum(s);
  if (lane == 0) atomicAdd(&AC[3], (double)s);
}

// K6: combine
__global__ void k_final(const double* __restrict__ AC, float* __restrict__ out) {
  if (threadIdx.x == 0 && blockIdx.x == 0) {
    double stab   = AC[0] / ((double)BATCH * (double)NUM_LABELS);
    double center = AC[1] / ((double)BATCH * (double)HIDDEN);
    double brier  = AC[2] / (double)BATCH;
    double sep    = AC[3] - (double)NUM_LABELS;
    double total  = 1.0 * stab + 0.1 * center + 0.01 * sep + 1.0 * brier;
    out[0] = (float)total; out[1] = (float)stab; out[2] = (float)center;
    out[3] = (float)sep;   out[4] = (float)brier;
  }
}

extern "C" void kernel_launch(void* const* d_in, const int* in_sizes, int n_in,
                              void* d_out, int out_size, void* d_ws, size_t ws_size,
                              hipStream_t stream) {
  const float* cls     = (const float*)d_in[0];
  const float* logits  = (const float*)d_in[1];
  const int*   labels  = (const int*)d_in[2];
  const float* W       = (const float*)d_in[3];
  const float* bias    = (const float*)d_in[4];
  const float* noise   = (const float*)d_in[5];
  const float* centers = (const float*)d_in[6];
  float* out = (float*)d_out;

  // workspace carve-up
  double*   AC     = (double*)d_ws;
  float*    counts = (float*)((char*)d_ws + 64);
  float*    sq     = counts + 1024;
  float*    sums   = sq + 1024;
  float*    cent   = sums + NUM_LABELS * HIDDEN;
  _Float16* Wt     = (_Float16*)(cent + NUM_LABELS * HIDDEN);

  (void)in_sizes; (void)n_in; (void)out_size; (void)ws_size;

  k_zero<<<(ZERO_FLOATS + 255) / 256, 256, 0, stream>>>((float*)d_ws);
  k_segsum<<<(BATCH * (HIDDEN / 4)) / 256, 256, 0, stream>>>(cls, labels, sums, counts);
  k_centers<<<(NUM_LABELS * HIDDEN) / 256, 256, 0, stream>>>(sums, counts, centers, cent, sq);
  k_wt<<<(NPAD * HIDDEN) / 256, 256, 0, stream>>>(W, Wt);
  k_gemm_stab<<<BATCH / MROWS, 256, 0, stream>>>(cls, noise, labels, logits, bias, Wt, cent, AC);
  k_brier<<<BATCH, 256, 0, stream>>>(logits, labels, AC);
  k_sep<<<dim3(NTILES, NTILES), 32, 0, stream>>>(cent, sq, AC);
  k_final<<<1, 1, 0, stream>>>(AC, out);
}
